// DecoderBlock_34479997452855
// MI455X (gfx1250) — matrix operand order repaired and checked by reading, not compiled, not run on hardware
//
#include <hip/hip_runtime.h>
#include <cstdint>
#include <cstddef>

// Problem constants (match the reference).
#define BB   2
#define TT   2048
#define EE   1024
#define HH   16
#define HDIM 64
#define FFD  4096
#define MROWS (BB * TT)   // 4096

typedef __attribute__((ext_vector_type(16))) __bf16 v16bf;
typedef __attribute__((ext_vector_type(8)))  float  v8f;
typedef unsigned short ushort_t;
typedef int v4i __attribute__((vector_size(16)));
typedef __attribute__((address_space(1))) v4i* gptr_v4i;
typedef __attribute__((address_space(3))) v4i* lptr_v4i;

#if __has_builtin(__builtin_amdgcn_global_load_async_to_lds_b128)
#define HAVE_ASYNC_LDS 1
#else
#define HAVE_ASYNC_LDS 0
#endif

static __device__ __forceinline__ void wait_asynccnt0() {
#if __has_builtin(__builtin_amdgcn_s_wait_asynccnt)
    __builtin_amdgcn_s_wait_asynccnt(0);
#else
    asm volatile("s_wait_asynccnt 0x0" ::: "memory");
#endif
}

union Frag {
    ushort_t s[16];
    uint4    q[2];
    v16bf    v;
};

static __device__ __forceinline__ ushort_t f2bf(float f) {
    union { float f; unsigned int u; } c;
    c.f = f;
    unsigned int u = c.u;
    unsigned int r = u + 0x7fffu + ((u >> 16) & 1u);   // round-to-nearest-even
    return (ushort_t)(r >> 16);
}

static __device__ __forceinline__ v8f wmma_bf16(const Frag& a, const Frag& b, v8f c) {
    return __builtin_amdgcn_wmma_f32_16x16x32_bf16(false, a.v, false, b.v,
                                                   (short)0, c, false, false);
}

// ---------------------------------------------------------------------------
// Weight convert + transpose: fp32 [K,N] row-major -> bf16 [N,K] row-major.
// ---------------------------------------------------------------------------
__global__ __launch_bounds__(256)
void k_wT(const float* __restrict__ W, ushort_t* __restrict__ out, int Kd, int Nd) {
    int idx = blockIdx.x * 256 + threadIdx.x;     // idx = k*Nd + n (coalesced read)
    int k = idx / Nd;
    int n = idx % Nd;
    out[(size_t)n * Kd + k] = f2bf(W[idx]);
}

// Wq/Wk/Wv: fp32 [H,E,HD] -> bf16 [H*HD, E] (transposed: out[n][e], n = h*HD+d)
__global__ __launch_bounds__(256)
void k_wqkvT(const float* __restrict__ W, ushort_t* __restrict__ out) {
    int idx = blockIdx.x * 256 + threadIdx.x;     // ((h*E + e)*HD + d)
    int d = idx % HDIM;
    int e = (idx / HDIM) % EE;
    int h = idx / (HDIM * EE);
    out[(size_t)(h * HDIM + d) * EE + e] = f2bf(W[idx]);
}

// ---------------------------------------------------------------------------
// LayerNorm (row = 1024 floats) -> bf16 row.
// ---------------------------------------------------------------------------
__global__ __launch_bounds__(256)
void k_ln_bf16(const float* __restrict__ x, const float* __restrict__ g,
               const float* __restrict__ be, ushort_t* __restrict__ out) {
    __shared__ float s1[256];
    __shared__ float s2[256];
    const int row = blockIdx.x;
    const float* xr = x + (size_t)row * EE;
    float v[4];
    float a = 0.f, sq = 0.f;
#pragma unroll
    for (int i = 0; i < 4; i++) {
        v[i] = xr[threadIdx.x + i * 256];
        a  += v[i];
        sq += v[i] * v[i];
    }
    s1[threadIdx.x] = a;
    s2[threadIdx.x] = sq;
    __syncthreads();
    for (int st = 128; st > 0; st >>= 1) {
        if (threadIdx.x < st) {
            s1[threadIdx.x] += s1[threadIdx.x + st];
            s2[threadIdx.x] += s2[threadIdx.x + st];
        }
        __syncthreads();
    }
    const float mu  = s1[0] * (1.0f / EE);
    const float var = s2[0] * (1.0f / EE) - mu * mu;
    const float rs  = rsqrtf(var + 1e-5f);
    ushort_t* orow = out + (size_t)row * EE;
#pragma unroll
    for (int i = 0; i < 4; i++) {
        int c = threadIdx.x + i * 256;
        orow[c] = f2bf((v[i] - mu) * rs * g[c] + be[c]);
    }
}

// ---------------------------------------------------------------------------
// Generic bf16 WMMA GEMM: C[M,N] = A[M,K] * Bt[N,K]^T  (+bias)(+resid)(relu)
// Wave computes a 16x64 tile (4 x v_wmma_f32_16x16x32_bf16 per K-step of 32).
// Outputs fp32 and/or bf16.
// ---------------------------------------------------------------------------
__global__ __launch_bounds__(256)
void k_gemm_bf16(const ushort_t* __restrict__ A, const ushort_t* __restrict__ Bt,
                 int M, int N, int K,
                 const float* __restrict__ bias,
                 const float* __restrict__ resid,
                 float* __restrict__ outF,
                 ushort_t* __restrict__ outB,
                 int relu) {
    const int lane = threadIdx.x & 31;
    const int wid  = (blockIdx.x * blockDim.x + threadIdx.x) >> 5;
    const int tilesN = N >> 6;
    const int m0 = (wid / tilesN) << 4;
    const int n0 = (wid % tilesN) << 6;
    const int half = lane >> 4;
    const int l15  = lane & 15;

    v8f c0 = {}, c1 = {}, c2 = {}, c3 = {};
    const ushort_t* arow = A  + (size_t)(m0 + l15) * K;
    const ushort_t* brow = Bt + (size_t)(n0 + l15) * K;

    for (int k = 0; k < K; k += 32) {
        // WGP-scope prefetch of the next chunk of the A and B streams.
        __builtin_prefetch(arow + k + 256, 0, 3);
        __builtin_prefetch(brow + k + 256, 0, 3);
        Frag a;
        a.q[0] = *(const uint4*)(arow + k + half * 8);
        a.q[1] = *(const uint4*)(arow + k + 16 + half * 8);
        Frag b;
        const ushort_t* bp;
        bp = brow + k + half * 8;
        b.q[0] = *(const uint4*)(bp); b.q[1] = *(const uint4*)(bp + 16);
        c0 = wmma_bf16(a, b, c0);
        bp = brow + (size_t)16 * K + k + half * 8;
        b.q[0] = *(const uint4*)(bp); b.q[1] = *(const uint4*)(bp + 16);
        c1 = wmma_bf16(a, b, c1);
        bp = brow + (size_t)32 * K + k + half * 8;
        b.q[0] = *(const uint4*)(bp); b.q[1] = *(const uint4*)(bp + 16);
        c2 = wmma_bf16(a, b, c2);
        bp = brow + (size_t)48 * K + k + half * 8;
        b.q[0] = *(const uint4*)(bp); b.q[1] = *(const uint4*)(bp + 16);
        c3 = wmma_bf16(a, b, c3);
    }

#pragma unroll
    for (int j = 0; j < 4; j++) {
        v8f c = (j == 0) ? c0 : (j == 1) ? c1 : (j == 2) ? c2 : c3;
        const int col = n0 + j * 16 + l15;
        const float bi = bias ? bias[col] : 0.0f;
#pragma unroll
        for (int r = 0; r < 8; r++) {
            const int row = m0 + r + half * 8;
            float val = c[r] + bi;
            if (relu) val = fmaxf(val, 0.0f);
            const size_t o = (size_t)row * N + col;
            if (resid) val += resid[o];
            if (outF) outF[o] = val;
            if (outB) outB[o] = f2bf(val);
        }
    }
}

// ---------------------------------------------------------------------------
// Flash attention: causal, scale = E^-0.5 = 1/32.
// Block = (b, h, 128 q-rows); 8 waves x 16 q-rows each.
// K tile staged via GLOBAL_LOAD_ASYNC_TO_LDS (ASYNCcnt path); V staged
// transposed through VGPRs; online softmax; P transposed via LDS.
// ---------------------------------------------------------------------------
__global__ __launch_bounds__(256)
void k_attn(const ushort_t* __restrict__ Q, const ushort_t* __restrict__ Kg,
            const ushort_t* __restrict__ Vg, ushort_t* __restrict__ Ob) {
    __shared__ ushort_t Kl[32 * 64];      // [key][d]
    __shared__ ushort_t Vt[64 * 32];      // [d][key]
    __shared__ ushort_t Pl[8 * 16 * 32];  // per-wave P tile [row][key]

    const int qtiles = TT / 128;          // 16
    const int bh = blockIdx.x / qtiles;
    const int qt = blockIdx.x % qtiles;
    const int b  = bh / HH;
    const int h  = bh % HH;
    const size_t rowbase = (size_t)b * TT;
    const int cbase = h * HDIM;

    const int tid  = threadIdx.x;
    const int lane = tid & 31;
    const int wv   = tid >> 5;
    const int half = lane >> 4;
    const int l15  = lane & 15;

    const int q0b = qt * 128;
    const int q0w = q0b + wv * 16;

    // Q fragments (A-layout), d-chunks [0,32) and [32,64)
    Frag aq0, aq1;
    {
        const ushort_t* qp = Q + (rowbase + q0w + l15) * EE + cbase;
        aq0.q[0] = *(const uint4*)(qp + half * 8);
        aq0.q[1] = *(const uint4*)(qp + 16 + half * 8);
        aq1.q[0] = *(const uint4*)(qp + 32 + half * 8);
        aq1.q[1] = *(const uint4*)(qp + 48 + half * 8);
    }

    float mrow[8], lrow[8];
#pragma unroll
    for (int r = 0; r < 8; r++) { mrow[r] = -3.0e38f; lrow[r] = 0.0f; }
    v8f o0 = {}, o1 = {}, o2 = {}, o3 = {};

    const float sc = 0.03125f;            // 1024^-0.5
    const int kend = q0b + 128;           // causal bound for this block

    for (int kb = 0; kb < kend; kb += 32) {
        __syncthreads();
        {   // cooperative K/V stage: thread -> 8 contiguous bf16 (one b128)
            const int key = tid >> 3;           // 0..31
            const int d   = (tid & 7) * 8;      // 0,8,...,56
            const size_t grow = (rowbase + kb + key) * EE + cbase + d;
#if HAVE_ASYNC_LDS
            // CDNA5 async DMA: memory -> LDS, tracked by ASYNCcnt.
            __builtin_amdgcn_global_load_async_to_lds_b128(
                (gptr_v4i)(Kg + grow),
                (lptr_v4i)(&Kl[key * 64 + d]),
                0, 0);
#else
            uint4 kv4 = *(const uint4*)(Kg + grow);
            *(uint4*)(&Kl[key * 64 + d]) = kv4;
#endif
            uint4 vv4 = *(const uint4*)(Vg + grow);
            unsigned int ws[4] = { vv4.x, vv4.y, vv4.z, vv4.w };
#pragma unroll
            for (int j = 0; j < 4; j++) {
                Vt[(d + 2 * j + 0) * 32 + key] = (ushort_t)(ws[j] & 0xffffu);
                Vt[(d + 2 * j + 1) * 32 + key] = (ushort_t)(ws[j] >> 16);
            }
#if HAVE_ASYNC_LDS
            wait_asynccnt0();
#endif
        }
        __syncthreads();

        // S = Q * K^T for keys [kb, kb+32): two 16x16 C tiles
        v8f s0 = {}, s1 = {};
        {
            Frag bk;
            const ushort_t* kp0 = &Kl[l15 * 64 + half * 8];
            bk.q[0] = *(const uint4*)(kp0);      bk.q[1] = *(const uint4*)(kp0 + 16);
            s0 = wmma_bf16(aq0, bk, s0);
            bk.q[0] = *(const uint4*)(kp0 + 32); bk.q[1] = *(const uint4*)(kp0 + 48);
            s0 = wmma_bf16(aq1, bk, s0);
            const ushort_t* kp1 = &Kl[(16 + l15) * 64 + half * 8];
            bk.q[0] = *(const uint4*)(kp1);      bk.q[1] = *(const uint4*)(kp1 + 16);
            s1 = wmma_bf16(aq0, bk, s1);
            bk.q[0] = *(const uint4*)(kp1 + 32); bk.q[1] = *(const uint4*)(kp1 + 48);
            s1 = wmma_bf16(aq1, bk, s1);
        }

        // Online softmax (per-row stats live replicated across 16-lane halves)
        const int key0 = kb + l15;
        const int key1 = kb + 16 + l15;
        float scl[8], p0v[8], p1v[8];
#pragma unroll
        for (int r = 0; r < 8; r++) {
            const int row = q0w + r + half * 8;
            float v0 = (key0 <= row) ? s0[r] * sc : -3.0e38f;
            float v1 = (key1 <= row) ? s1[r] * sc : -3.0e38f;
            float mx = fmaxf(v0, v1);
            mx = fmaxf(mx, __shfl_xor(mx, 1));
            mx = fmaxf(mx, __shfl_xor(mx, 2));
            mx = fmaxf(mx, __shfl_xor(mx, 4));
            mx = fmaxf(mx, __shfl_xor(mx, 8));
            const float mn = fmaxf(mrow[r], mx);
            const float e  = __expf(mrow[r] - mn);
            const float p0 = __expf(v0 - mn);
            const float p1 = __expf(v1 - mn);
            float ps = p0 + p1;
            ps += __shfl_xor(ps, 1);
            ps += __shfl_xor(ps, 2);
            ps += __shfl_xor(ps, 4);
            ps += __shfl_xor(ps, 8);
            lrow[r] = lrow[r] * e + ps;
            mrow[r] = mn;
            scl[r] = e;
            p0v[r] = p0;
            p1v[r] = p1;
        }
#pragma unroll
        for (int r = 0; r < 8; r++) {
            o0[r] *= scl[r]; o1[r] *= scl[r]; o2[r] *= scl[r]; o3[r] *= scl[r];
            // P: C-layout -> LDS (per-wave region)
            Pl[wv * 512 + (r + half * 8) * 32 + l15]      = f2bf(p0v[r]);
            Pl[wv * 512 + (r + half * 8) * 32 + 16 + l15] = f2bf(p1v[r]);
        }
        __syncthreads();   // make P visible across the wave's lanes

        // P as A-fragment (16 rows x 32 keys)
        Frag pa;
        const ushort_t* pp = &Pl[wv * 512 + l15 * 32];
        pa.q[0] = *(const uint4*)(pp + half * 8);
        pa.q[1] = *(const uint4*)(pp + 16 + half * 8);

        // O += P * V  (4 n-tiles of 16 d-cols)
        Frag bv;
        const ushort_t* vp;
        vp = &Vt[(0 + l15) * 32 + half * 8];
        bv.q[0] = *(const uint4*)(vp); bv.q[1] = *(const uint4*)(vp + 16);
        o0 = wmma_bf16(pa, bv, o0);
        vp = &Vt[(16 + l15) * 32 + half * 8];
        bv.q[0] = *(const uint4*)(vp); bv.q[1] = *(const uint4*)(vp + 16);
        o1 = wmma_bf16(pa, bv, o1);
        vp = &Vt[(32 + l15) * 32 + half * 8];
        bv.q[0] = *(const uint4*)(vp); bv.q[1] = *(const uint4*)(vp + 16);
        o2 = wmma_bf16(pa, bv, o2);
        vp = &Vt[(48 + l15) * 32 + half * 8];
        bv.q[0] = *(const uint4*)(vp); bv.q[1] = *(const uint4*)(vp + 16);
        o3 = wmma_bf16(pa, bv, o3);
    }

    // Normalize and emit bf16 (concat-head layout [B*T, E])
#pragma unroll
    for (int r = 0; r < 8; r++) {
        const float inv = 1.0f / lrow[r];
        const int row = q0w + r + half * 8;
        ushort_t* op = Ob + (rowbase + row) * EE + cbase;
        op[l15]      = f2bf(o0[r] * inv);
        op[16 + l15] = f2bf(o1[r] * inv);
        op[32 + l15] = f2bf(o2[r] * inv);
        op[48 + l15] = f2bf(o3[r] * inv);
    }
}

// ---------------------------------------------------------------------------
// Host-side orchestration.
// ---------------------------------------------------------------------------
extern "C" void kernel_launch(void* const* d_in, const int* in_sizes, int n_in,
                              void* d_out, int out_size, void* d_ws, size_t ws_size,
                              hipStream_t stream) {
    const float* x     = (const float*)d_in[0];
    const float* Wq    = (const float*)d_in[1];
    const float* Wk    = (const float*)d_in[2];
    const float* Wv    = (const float*)d_in[3];
    const float* Wproj = (const float*)d_in[4];
    const float* bproj = (const float*)d_in[5];
    const float* W1    = (const float*)d_in[6];
    const float* b1    = (const float*)d_in[7];
    const float* W2    = (const float*)d_in[8];
    const float* b2    = (const float*)d_in[9];
    const float* g1    = (const float*)d_in[10];
    const float* be1   = (const float*)d_in[11];
    const float* g2    = (const float*)d_in[12];
    const float* be2   = (const float*)d_in[13];
    float* out = (float*)d_out;

    char* ws = (char*)d_ws;
    size_t off = 0;
    auto alloc = [&](size_t bytes) -> void* {
        void* p = ws + off;
        off = (off + bytes + 255) & ~(size_t)255;
        return p;
    };

    const size_t M = MROWS;
    ushort_t* wqT    = (ushort_t*)alloc((size_t)EE * EE * 2);
    ushort_t* wkT    = (ushort_t*)alloc((size_t)EE * EE * 2);
    ushort_t* wvT    = (ushort_t*)alloc((size_t)EE * EE * 2);
    ushort_t* wprojT = (ushort_t*)alloc((size_t)EE * EE * 2);
    ushort_t* w1T    = (ushort_t*)alloc((size_t)EE * FFD * 2);
    ushort_t* w2T    = (ushort_t*)alloc((size_t)FFD * EE * 2);
    ushort_t* hln    = (ushort_t*)alloc(M * EE * 2);
    ushort_t* qb     = (ushort_t*)alloc(M * EE * 2);
    ushort_t* kb     = (ushort_t*)alloc(M * EE * 2);
    ushort_t* vb     = (ushort_t*)alloc(M * EE * 2);
    ushort_t* attn   = (ushort_t*)alloc(M * EE * 2);
    float*    x2     = (float*)   alloc(M * EE * 4);
    ushort_t* h2     = (ushort_t*)alloc(M * EE * 2);
    ushort_t* f1     = (ushort_t*)alloc(M * FFD * 2);

    // 1) weight conversion (bf16, transposed)
    k_wqkvT<<<(HH * EE * HDIM) / 256, 256, 0, stream>>>(Wq, wqT);
    k_wqkvT<<<(HH * EE * HDIM) / 256, 256, 0, stream>>>(Wk, wkT);
    k_wqkvT<<<(HH * EE * HDIM) / 256, 256, 0, stream>>>(Wv, wvT);
    k_wT<<<(EE * EE)  / 256, 256, 0, stream>>>(Wproj, wprojT, EE, EE);
    k_wT<<<(EE * FFD) / 256, 256, 0, stream>>>(W1, w1T, EE, FFD);
    k_wT<<<(FFD * EE) / 256, 256, 0, stream>>>(W2, w2T, FFD, EE);

    // 2) LN1 -> bf16
    k_ln_bf16<<<(int)M, 256, 0, stream>>>(x, g1, be1, hln);

    // 3) QKV projections (WMMA GEMMs)
    const int gemmE  = (int)((M / 16) * (EE  / 64) / 8);   // 512 blocks
    const int gemmFF = (int)((M / 16) * (FFD / 64) / 8);   // 2048 blocks
    k_gemm_bf16<<<gemmE, 256, 0, stream>>>(hln, wqT, (int)M, EE, EE,
                                           nullptr, nullptr, nullptr, qb, 0);
    k_gemm_bf16<<<gemmE, 256, 0, stream>>>(hln, wkT, (int)M, EE, EE,
                                           nullptr, nullptr, nullptr, kb, 0);
    k_gemm_bf16<<<gemmE, 256, 0, stream>>>(hln, wvT, (int)M, EE, EE,
                                           nullptr, nullptr, nullptr, vb, 0);

    // 4) causal flash attention
    k_attn<<<BB * HH * (TT / 128), 256, 0, stream>>>(qb, kb, vb, attn);

    // 5) out-proj + bias + residual -> x2 (fp32)
    k_gemm_bf16<<<gemmE, 256, 0, stream>>>(attn, wprojT, (int)M, EE, EE,
                                           bproj, x, x2, nullptr, 0);

    // 6) LN2 -> bf16
    k_ln_bf16<<<(int)M, 256, 0, stream>>>(x2, g2, be2, h2);

    // 7) FFN: relu(h2 @ W1 + b1) -> bf16
    k_gemm_bf16<<<gemmFF, 256, 0, stream>>>(h2, w1T, (int)M, FFD, EE,
                                            b1, nullptr, nullptr, f1, 1);

    // 8) FFN: f1 @ W2 + b2 + x2 -> d_out (fp32)
    k_gemm_bf16<<<gemmE, 256, 0, stream>>>(f1, w2T, (int)M, EE, FFD,
                                           b2, x2, out, nullptr, 0);
}
